// EulerGCN_6365141532815
// MI455X (gfx1250) — matrix operand order repaired and checked by reading, not compiled, not run on hardware
//
#include <hip/hip_runtime.h>
#include <hip/hip_bf16.h>
#include <stdint.h>

typedef __attribute__((ext_vector_type(16))) __bf16 v16bf;
typedef __attribute__((ext_vector_type(8)))  __bf16 v8bf;
typedef __attribute__((ext_vector_type(8)))  float  v8f;

#define HDIM 64
#define XDIM 128
#define ZDIM 32
#define TSTEPS 8

// ---------------- WMMA fragment helpers ----------------

static __device__ __forceinline__ v8f zero8() {
  v8f v;
#pragma unroll
  for (int i = 0; i < 8; ++i) v[i] = 0.0f;
  return v;
}

// A fragment 16x32 (MxK) from row-major bf16 activations: two 16B contiguous loads.
static __device__ __forceinline__ v16bf load_a_bf(const __bf16* __restrict__ A, int lda,
                                                  int row0, int k0, int nrows) {
  int lane = threadIdx.x & 31;
  int h = lane >> 4, m = lane & 15;
  int r = row0 + m; if (r >= nrows) r = nrows - 1;     // clamp, EXEC stays uniform
  const __bf16* p = A + (long)r * lda + k0 + 16 * h;   // 16B aligned
  v8bf lo = *(const v8bf*)p;          // K = k0+16h .. +7
  v8bf hi = *(const v8bf*)(p + 8);    // K = k0+16h+8 .. +7
  v16bf a;
#pragma unroll
  for (int j = 0; j < 8; ++j) { a[j] = lo[j]; a[j + 8] = hi[j]; }
  return a;
}

// B fragment from pre-packed weight tiles: 32 contiguous bytes per lane (2x b128).
static __device__ __forceinline__ v16bf load_pb(const __bf16* __restrict__ PB, int tile) {
  int lane = threadIdx.x & 31;
  return *(const v16bf*)(PB + ((long)tile * 32 + lane) * 16);
}

static __device__ __forceinline__ v8f wmma_bf16(v16bf a, v16bf b, v8f c) {
  return __builtin_amdgcn_wmma_f32_16x16x32_bf16(false, a, false, b, (short)0, c, false, false);
}

// ---------------- weight packing (once per launch; tiny) ----------------
// Packed layout: PB[(kt*nct + ct)*512 + lane*16 + j] = bf16(B[kt*32 + 16*(lane>>4) + j][ct*16 + (lane&15)])

__global__ void k_pack_b(const float* __restrict__ W, int K, int Ncols,
                         __bf16* __restrict__ PB) {           // W row-major [K][Ncols]
  int idx = blockIdx.x * blockDim.x + threadIdx.x;
  int total = (K >> 5) * (Ncols >> 4) * 512;
  if (idx >= total) return;
  int j = idx & 15, lane = (idx >> 4) & 31, tile = idx >> 9;
  int nct = Ncols >> 4;
  int kt = tile / nct, ct = tile % nct;
  int k = kt * 32 + 16 * (lane >> 4) + j, c = ct * 16 + (lane & 15);
  PB[idx] = (__bf16)W[(long)k * Ncols + c];
}

__global__ void k_pack_bt(const float* __restrict__ W, int K, int Ncols,
                          __bf16* __restrict__ PB) {          // W row-major [Ncols][K] (x@W.T)
  int idx = blockIdx.x * blockDim.x + threadIdx.x;
  int total = (K >> 5) * (Ncols >> 4) * 512;
  if (idx >= total) return;
  int j = idx & 15, lane = (idx >> 4) & 31, tile = idx >> 9;
  int nct = Ncols >> 4;
  int kt = tile / nct, ct = tile % nct;
  int k = kt * 32 + 16 * (lane >> 4) + j, c = ct * 16 + (lane & 15);
  PB[idx] = (__bf16)W[(long)c * K + k];
}

// ---------------- GEMM kernels ----------------

// xw1[N,64] = x_bf[N,128] @ W1  (packed)
__global__ void k_gemm_xw1(const __bf16* __restrict__ x_bf, const __bf16* __restrict__ PB,
                           float* __restrict__ out, int N) {
  int wid = (blockIdx.x * blockDim.x + threadIdx.x) >> 5;
  int ntiles = (N + 15) >> 4;
  if (wid >= ntiles) return;
  int row0 = wid << 4;
  v8f acc[4];
#pragma unroll
  for (int c = 0; c < 4; ++c) acc[c] = zero8();
#pragma unroll
  for (int kt = 0; kt < 4; ++kt) {
    v16bf a = load_a_bf(x_bf, XDIM, row0, kt * 32, N);
#pragma unroll
    for (int c = 0; c < 4; ++c)
      acc[c] = wmma_bf16(a, load_pb(PB, kt * 4 + c), acc[c]);
  }
  int lane = threadIdx.x & 31, h = lane >> 4, n = lane & 15;
#pragma unroll
  for (int c = 0; c < 4; ++c)
#pragma unroll
    for (int i = 0; i < 8; ++i) {
      int r = row0 + i + 8 * h;
      if (r < N) out[(long)r * HDIM + c * 16 + n] = acc[c][i];
    }
}

// out[N,64] = act_bf[N,64] @ W2 (packed)
__global__ void k_gemm_h64(const __bf16* __restrict__ A, const __bf16* __restrict__ PB,
                           float* __restrict__ out, int N) {
  int wid = (blockIdx.x * blockDim.x + threadIdx.x) >> 5;
  int ntiles = (N + 15) >> 4;
  if (wid >= ntiles) return;
  int row0 = wid << 4;
  v8f acc[4];
#pragma unroll
  for (int c = 0; c < 4; ++c) acc[c] = zero8();
#pragma unroll
  for (int kt = 0; kt < 2; ++kt) {
    v16bf a = load_a_bf(A, HDIM, row0, kt * 32, N);
#pragma unroll
    for (int c = 0; c < 4; ++c)
      acc[c] = wmma_bf16(a, load_pb(PB, kt * 4 + c), acc[c]);
  }
  int lane = threadIdx.x & 31, h = lane >> 4, n = lane & 15;
#pragma unroll
  for (int c = 0; c < 4; ++c)
#pragma unroll
    for (int i = 0; i < 8; ++i) {
      int r = row0 + i + 8 * h;
      if (r < N) out[(long)r * HDIM + c * 16 + n] = acc[c][i];
    }
}

// Fused GRU step: one wave per 16x16 tile of h_new; 6 WMMAs per k-step.
__global__ void k_gru(const __bf16* __restrict__ xg_bf, const __bf16* __restrict__ hold_bf,
                      const float* __restrict__ hold_f,
                      const __bf16* __restrict__ PBih, const __bf16* __restrict__ PBhh,
                      const float* __restrict__ bih, const float* __restrict__ bhh,
                      float* __restrict__ hnew_f, __bf16* __restrict__ hnew_bf, int N) {
  int wid = (blockIdx.x * blockDim.x + threadIdx.x) >> 5;
  int ntiles = ((N + 15) >> 4) * 4;
  if (wid >= ntiles) return;
  int row0 = (wid >> 2) << 4, ct = wid & 3, col0 = ct << 4;
  v8f ir = zero8(), iz = zero8(), in_ = zero8();
  v8f hr = zero8(), hz = zero8(), hn = zero8();
#pragma unroll
  for (int kt = 0; kt < 2; ++kt) {
    v16bf ax = load_a_bf(xg_bf,   HDIM, row0, kt * 32, N);
    v16bf ah = load_a_bf(hold_bf, HDIM, row0, kt * 32, N);
    int tb = kt * 12 + ct;                 // 12 col-tiles across 3H=192
    ir  = wmma_bf16(ax, load_pb(PBih, tb),     ir);
    iz  = wmma_bf16(ax, load_pb(PBih, tb + 4), iz);
    in_ = wmma_bf16(ax, load_pb(PBih, tb + 8), in_);
    hr  = wmma_bf16(ah, load_pb(PBhh, tb),     hr);
    hz  = wmma_bf16(ah, load_pb(PBhh, tb + 4), hz);
    hn  = wmma_bf16(ah, load_pb(PBhh, tb + 8), hn);
  }
  int lane = threadIdx.x & 31, hh = lane >> 4, n = lane & 15;
  int col = col0 + n;
  float b_r  = bih[col]        + bhh[col];
  float b_z  = bih[HDIM + col] + bhh[HDIM + col];
  float bi_n = bih[2 * HDIM + col];
  float bh_n = bhh[2 * HDIM + col];
#pragma unroll
  for (int i = 0; i < 8; ++i) {
    int r = row0 + i + 8 * hh;
    if (r >= N) continue;
    float vr = ir[i] + hr[i] + b_r;
    float vz = iz[i] + hz[i] + b_z;
    float rr = 1.0f / (1.0f + __expf(-vr));
    float zz = 1.0f / (1.0f + __expf(-vz));
    float vn = tanhf((in_[i] + bi_n) + rr * (hn[i] + bh_n));
    float hp = hold_f[(long)r * HDIM + col];
    float hv = (1.0f - zz) * vn + zz * hp;
    hnew_f[(long)r * HDIM + col]  = hv;
    hnew_bf[(long)r * HDIM + col] = (__bf16)hv;
  }
}

// out[N,32] = h_bf[N,64] @ Wlin.T (packed) + blin
__global__ void k_linear(const __bf16* __restrict__ h_bf, const __bf16* __restrict__ PB,
                         const float* __restrict__ blin, float* __restrict__ out, int N) {
  int wid = (blockIdx.x * blockDim.x + threadIdx.x) >> 5;
  int ntiles = ((N + 15) >> 4) * 2;
  if (wid >= ntiles) return;
  int row0 = (wid >> 1) << 4, ct = wid & 1, col0 = ct << 4;
  v8f acc = zero8();
#pragma unroll
  for (int kt = 0; kt < 2; ++kt)
    acc = wmma_bf16(load_a_bf(h_bf, HDIM, row0, kt * 32, N),
                    load_pb(PB, kt * 2 + ct), acc);
  int lane = threadIdx.x & 31, h = lane >> 4, n = lane & 15;
  float bb = blin[col0 + n];
#pragma unroll
  for (int i = 0; i < 8; ++i) {
    int r = row0 + i + 8 * h;
    if (r < N) out[(long)r * ZDIM + col0 + n] = acc[i] + bb;
  }
}

// ---------------- graph / elementwise kernels ----------------

__global__ void k_zero(float* __restrict__ p, long n) {
  long i = (long)blockIdx.x * blockDim.x + threadIdx.x;
  if (i < n) p[i] = 0.0f;
}

__global__ void k_zero_bf(__bf16* __restrict__ p, long n) {
  long i = (long)blockIdx.x * blockDim.x + threadIdx.x;
  if (i < n) p[i] = (__bf16)0.0f;
}

__global__ void k_cvt_bf(const float* __restrict__ in, __bf16* __restrict__ out, long n) {
  long i = (long)blockIdx.x * blockDim.x + threadIdx.x;
  if (i < n) out[i] = (__bf16)in[i];
}

__global__ void k_deg_init(float* __restrict__ deg, int N) {
  int i = blockIdx.x * blockDim.x + threadIdx.x;
  if (i < N) deg[i] = 1.0f;                       // self-loop weight
}

__global__ void k_deg_acc(const int* __restrict__ dst, const float* __restrict__ ew,
                          float* __restrict__ deg, int E) {
  int e = blockIdx.x * blockDim.x + threadIdx.x;
  if (e < E) atomicAdd(&deg[dst[e]], ew[e]);
}

__global__ void k_dinv(float* __restrict__ deg, int N) {
  int i = blockIdx.x * blockDim.x + threadIdx.x;
  if (i < N) { float d = deg[i]; deg[i] = d > 0.0f ? rsqrtf(d) : 0.0f; }
}

__global__ void k_norm(const int* __restrict__ src, const int* __restrict__ dst,
                       const float* __restrict__ ew, const float* __restrict__ dinv,
                       float* __restrict__ nrm, int E) {
  int e = blockIdx.x * blockDim.x + threadIdx.x;
  if (e < E) nrm[e] = dinv[src[e]] * ew[e] * dinv[dst[e]];
}

// hout[n,:] = dinv[n]^2 * hin[n,:]   (self-loop term seeds the scatter target)
__global__ void k_self(const float* __restrict__ hin, const float* __restrict__ dinv,
                       float* __restrict__ hout, int N) {
  int t = blockIdx.x * blockDim.x + threadIdx.x;
  if (t >= N * (HDIM / 4)) return;
  int node = t >> 4, q = t & 15;
  float s = dinv[node]; s *= s;
  float4 v = ((const float4*)(hin + (long)node * HDIM))[q];
  float4 o; o.x = v.x * s; o.y = v.y * s; o.z = v.z * s; o.w = v.w * s;
  ((float4*)(hout + (long)node * HDIM))[q] = o;
}

// Edge-parallel scatter-add: hout[dst] += norm * hin[src]. 16 threads/edge, float4 each.
__global__ void k_scatter(const int* __restrict__ src, const int* __restrict__ dst,
                          const float* __restrict__ nrm, const float* __restrict__ hin,
                          float* __restrict__ hout, int E) {
  int t = blockIdx.x * blockDim.x + threadIdx.x;
  int e = t >> 4, q = t & 15;
  if (e >= E) return;
  if (q == 0 && e + 1 < E)
    __builtin_prefetch(hin + (long)src[e + 1] * HDIM, 0, 0);   // global_prefetch_b8
  float w = nrm[e];
  const float* ps = hin + (long)src[e] * HDIM + q * 4;
  float* pd = hout + (long)dst[e] * HDIM + q * 4;
  float4 v = *(const float4*)ps;
  atomicAdd(pd + 0, w * v.x);
  atomicAdd(pd + 1, w * v.y);
  atomicAdd(pd + 2, w * v.z);
  atomicAdd(pd + 3, w * v.w);
}

// relu(x+b1) -> bf16 (consumed only as GEMM A operand)
__global__ void k_bias_relu_bf(const float* __restrict__ in, const float* __restrict__ b,
                               __bf16* __restrict__ out, long n) {
  long i = (long)blockIdx.x * blockDim.x + threadIdx.x;
  if (i < n) { float v = in[i] + b[i & (HDIM - 1)]; out[i] = (__bf16)(v > 0.0f ? v : 0.0f); }
}

// tanh(x+b2) -> bf16 (consumed only as GEMM A operand)
__global__ void k_bias_tanh_bf(const float* __restrict__ in, const float* __restrict__ b,
                               __bf16* __restrict__ out, long n) {
  long i = (long)blockIdx.x * blockDim.x + threadIdx.x;
  if (i < n) out[i] = (__bf16)tanhf(in[i] + b[i & (HDIM - 1)]);
}

// ---------------- host orchestration ----------------

extern "C" void kernel_launch(void* const* d_in, const int* in_sizes, int n_in,
                              void* d_out, int out_size, void* d_ws, size_t ws_size,
                              hipStream_t stream) {
  const float* x    = (const float*)d_in[0];
  const int*   eis  = (const int*)  d_in[1];
  const float* ews  = (const float*)d_in[2];
  const float* W1   = (const float*)d_in[3];
  const float* b1   = (const float*)d_in[4];
  const float* W2   = (const float*)d_in[5];
  const float* b2   = (const float*)d_in[6];
  const float* Wih  = (const float*)d_in[7];
  const float* Whh  = (const float*)d_in[8];
  const float* bih  = (const float*)d_in[9];
  const float* bhh  = (const float*)d_in[10];
  const float* Wlin = (const float*)d_in[11];
  const float* blin = (const float*)d_in[12];

  const int N = in_sizes[0] / XDIM;
  const int E = in_sizes[1] / (2 * TSTEPS);
  float* out = (float*)d_out;
  const long nh = (long)N * HDIM;
  const long nx = (long)N * XDIM;

  // ---- workspace carving ----
  char* base = (char*)d_ws;
  float* xw1  = (float*)base; base += nh * 4;
  float* bufA = (float*)base; base += nh * 4;   // scatter target L1
  float* bufB = (float*)base; base += nh * 4;   // h1 @ W2
  float* bufC = (float*)base; base += nh * 4;   // embed (scatter target L2)
  float* hf0  = (float*)base; base += nh * 4;   // GRU state f32 (double buffer)
  float* hf1  = (float*)base; base += nh * 4;
  float* dinv = (float*)base; base += (long)N * 4;
  float* nrm  = (float*)base; base += (long)E * 4;
  base = (char*)(((uintptr_t)base + 31) & ~(uintptr_t)31);
  __bf16* x_bf   = (__bf16*)base; base += nx * 2;
  __bf16* act_bf = (__bf16*)base; base += nh * 2;   // relu output
  __bf16* xg_bf  = (__bf16*)base; base += nh * 2;   // tanh output (GRU input)
  __bf16* hbf0   = (__bf16*)base; base += nh * 2;   // GRU state bf16 mirror
  __bf16* hbf1   = (__bf16*)base; base += nh * 2;
  const int PW1_T  = 4 * 4 * 512;    // W1   packed: K=128, N=64
  const int PW2_T  = 2 * 4 * 512;    // W2   packed: K=64,  N=64
  const int PIH_T  = 2 * 12 * 512;   // WihT packed: K=64,  N=192
  const int PHH_T  = 2 * 12 * 512;
  const int PLN_T  = 2 * 2 * 512;    // WlinT packed: K=64, N=32
  __bf16* PBW1 = (__bf16*)base; base += (long)PW1_T * 2;
  __bf16* PBW2 = (__bf16*)base; base += (long)PW2_T * 2;
  __bf16* PBih = (__bf16*)base; base += (long)PIH_T * 2;
  __bf16* PBhh = (__bf16*)base; base += (long)PHH_T * 2;
  __bf16* PBln = (__bf16*)base; base += (long)PLN_T * 2;
  float* hf[2]   = {hf0, hf1};
  __bf16* hbf[2] = {hbf0, hbf1};

  const int TH = 256;
  const int rowTiles = (N + 15) / 16;
  const int gB_n   = (N + TH - 1) / TH;
  const int gB_e   = (E + TH - 1) / TH;
  const int gB_nh  = (int)((nh + TH - 1) / TH);
  const int gB_nx  = (int)((nx + TH - 1) / TH);
  const int gB_n4  = (N * 16 + TH - 1) / TH;
  const int gB_e16 = (int)(((long)E * 16 + TH - 1) / TH);
  const int gW1 = (rowTiles * 32 + TH - 1) / TH;
  const int gW4 = (rowTiles * 4 * 32 + TH - 1) / TH;
  const int gW2 = (rowTiles * 2 * 32 + TH - 1) / TH;

  // ---- one-time per launch: h0 = 0, x->bf16, pack all weights ----
  k_zero   <<<gB_nh, TH, 0, stream>>>(hf[0], nh);
  k_zero_bf<<<gB_nh, TH, 0, stream>>>(hbf[0], nh);
  k_cvt_bf <<<gB_nx, TH, 0, stream>>>(x, x_bf, nx);
  k_pack_b <<<(PW1_T + TH - 1) / TH, TH, 0, stream>>>(W1,   XDIM, HDIM,     PBW1);
  k_pack_b <<<(PW2_T + TH - 1) / TH, TH, 0, stream>>>(W2,   HDIM, HDIM,     PBW2);
  k_pack_bt<<<(PIH_T + TH - 1) / TH, TH, 0, stream>>>(Wih,  HDIM, 3 * HDIM, PBih);
  k_pack_bt<<<(PHH_T + TH - 1) / TH, TH, 0, stream>>>(Whh,  HDIM, 3 * HDIM, PBhh);
  k_pack_bt<<<(PLN_T + TH - 1) / TH, TH, 0, stream>>>(Wlin, HDIM, ZDIM,     PBln);

  k_gemm_xw1<<<gW1, TH, 0, stream>>>(x_bf, PBW1, xw1, N);             // WMMA

  for (int t = 0; t < TSTEPS; ++t) {
    const int*   src = eis + (long)t * 2 * E;
    const int*   dst = src + E;
    const float* ew  = ews + (long)t * E;

    k_deg_init<<<gB_n, TH, 0, stream>>>(dinv, N);
    k_deg_acc <<<gB_e, TH, 0, stream>>>(dst, ew, dinv, E);
    k_dinv    <<<gB_n, TH, 0, stream>>>(dinv, N);
    k_norm    <<<gB_e, TH, 0, stream>>>(src, dst, ew, dinv, nrm, E);

    // GCN layer 1: prop(xw1) -> relu(+b1) -> bf16
    k_self        <<<gB_n4,  TH, 0, stream>>>(xw1, dinv, bufA, N);
    k_scatter     <<<gB_e16, TH, 0, stream>>>(src, dst, nrm, xw1, bufA, E);
    k_bias_relu_bf<<<gB_nh,  TH, 0, stream>>>(bufA, b1, act_bf, nh);

    // GCN layer 2: (h1 @ W2) -> prop -> tanh(+b2) -> bf16
    k_gemm_h64    <<<gW1,    TH, 0, stream>>>(act_bf, PBW2, bufB, N); // WMMA
    k_self        <<<gB_n4,  TH, 0, stream>>>(bufB, dinv, bufC, N);
    k_scatter     <<<gB_e16, TH, 0, stream>>>(src, dst, nrm, bufB, bufC, E);
    k_bias_tanh_bf<<<gB_nh,  TH, 0, stream>>>(bufC, b2, xg_bf, nh);

    // GRU step (6 WMMAs/tile) + output projection (WMMA)
    k_gru   <<<gW4, TH, 0, stream>>>(xg_bf, hbf[t & 1], hf[t & 1], PBih, PBhh,
                                     bih, bhh, hf[(t + 1) & 1], hbf[(t + 1) & 1], N);
    k_linear<<<gW2, TH, 0, stream>>>(hbf[(t + 1) & 1], PBln, blin,
                                     out + (long)t * N * ZDIM, N);
  }
}
